// DebugNet_71347996721718
// MI455X (gfx1250) — matrix operand order repaired and checked by reading, not compiled, not run on hardware
//
#include <hip/hip_runtime.h>
#include <hip/hip_bf16.h>

typedef _Float16 f16;
typedef __attribute__((ext_vector_type(16))) _Float16 v16h;
typedef __attribute__((ext_vector_type(8)))  _Float16 v8h;
typedef __attribute__((ext_vector_type(8)))  float    v8f;

#define NB    8
#define NA    48
#define GF    256
#define NS    4096
#define NP    32
#define NE    5
#define OHN   100
#define KPAD  384
#define NBINS 65536
#define ROWS  (NB * NS)

// ---------------------------------------------------------------------------
// Repack fp32 weight (K x 256, row-major) into WMMA B-fragment order:
// out[((t*16 + j)*32 + lane)*16 + e] = W[t*32 + (e<8 ? 8*(lane/16)+e : 16+8*(lane/16)+e-8)][j*16 + lane%16]
// Rows >= K are zero-padded (combine_W: K=356 -> Kpad=384).
// ---------------------------------------------------------------------------
__global__ void pack_w_kernel(const float* __restrict__ W, int K, int Kpad,
                              f16* __restrict__ out) {
    int idx = blockIdx.x * 256 + threadIdx.x;
    int total = (Kpad / 32) * 16 * 32 * 16;
    if (idx >= total) return;
    int e    = idx & 15;
    int lane = (idx >> 4) & 31;
    int j    = (idx >> 9) & 15;
    int t    = idx >> 13;
    int h = lane >> 4, n = lane & 15;
    int krow = t * 32 + ((e < 8) ? (h * 8 + e) : (16 + h * 8 + (e - 8)));
    int col  = j * 16 + n;
    float v = (krow < K) ? W[krow * 256 + col] : 0.0f;
    out[idx] = (f16)v;
}

// ---------------------------------------------------------------------------
// Stage A: one block per (b,s).  Subset-weighted feature mean, instance norm,
// thermometer one-hot -> f16 activation row of width KPAD (=384).
// ---------------------------------------------------------------------------
__global__ __launch_bounds__(256) void stage_a_kernel(
    const float* __restrict__ feat,     // (B,A,GF)
    const float* __restrict__ mask,     // (B,A)
    const float* __restrict__ eloh,     // (B,A,E)
    const int*   __restrict__ subsets,  // (B,S,A)
    f16* __restrict__ act0)             // (ROWS, KPAD)
{
    int row = blockIdx.x;               // b*NS + s
    int b = row / NS;
    int tid = threadIdx.x;

    __shared__ float subf[NA], msk[NA], cnt[NE];
    __shared__ float red[256];
    __shared__ float s_size, s_mean, s_rstd;

    if (tid < NA) {
        subf[tid] = (float)subsets[(size_t)row * NA + tid];
        msk[tid]  = mask[b * NA + tid];
    }
    __syncthreads();

    if (tid < NE) {                      // counts = oh^T * subsets_f (unmasked)
        float c = 0.0f;
        for (int a = 0; a < NA; ++a) c += subf[a] * eloh[(b * NA + a) * NE + tid];
        cnt[tid] = c;
    }
    if (tid == NE) {                     // subset_size = sum(subsets*mask) + eps
        float sz = 0.0f;
        for (int a = 0; a < NA; ++a) sz += subf[a] * msk[a];
        s_size = sz + 1e-4f;
    }
    __syncthreads();

    // sws[g] = sum_a (subsets*mask)[a] * (feat*mask)[a][g]
    float sws = 0.0f;
    for (int a = 0; a < NA; ++a) {
        float w = subf[a] * msk[a] * msk[a];
        sws += w * feat[(size_t)(b * NA + a) * GF + tid];
    }
    float x = sws / s_size;

    // instance norm over GF=256
    red[tid] = x; __syncthreads();
    for (int off = 128; off > 0; off >>= 1) {
        if (tid < off) red[tid] += red[tid + off];
        __syncthreads();
    }
    if (tid == 0) s_mean = red[0] * (1.0f / 256.0f);
    __syncthreads();
    float d = x - s_mean;
    red[tid] = d * d; __syncthreads();
    for (int off = 128; off > 0; off >>= 1) {
        if (tid < off) red[tid] += red[tid + off];
        __syncthreads();
    }
    if (tid == 0) s_rstd = rsqrtf(red[0] * (1.0f / 256.0f) + 1e-5f);
    __syncthreads();

    f16* rp = act0 + (size_t)row * KPAD;
    rp[OHN + tid] = (f16)(d * s_rstd);
    if (tid < OHN) {                     // thermometer one-hot, 5 segments of 20
        int seg = tid / 20, o = tid % 20;
        rp[tid] = (f16)(((float)o < cnt[seg]) ? 1.0f : 0.0f);
    }
    if (tid < KPAD - OHN - GF) rp[OHN + GF + tid] = (f16)0.0f;   // zero pad
}

// ---------------------------------------------------------------------------
// WMMA GEMM: out(ROWS x 256) = act(ROWS x K, f16, stride lda) @ W(K x 256)
// Block = 256 threads = 8 waves; block tile 64 x 256; each wave: 16 x 128.
// mode: 0 relu->f32, 1 relu->f16, 2 leaky(0.01)->f16, 3 raw->f32
// ---------------------------------------------------------------------------
__global__ __launch_bounds__(256) void gemm_wmma_kernel(
    const f16* __restrict__ A, int lda,
    const f16* __restrict__ Bp, int Kt,
    const float* __restrict__ bias,
    void* __restrict__ outp, int mode)
{
    int w    = threadIdx.x >> 5;
    int lane = threadIdx.x & 31;
    int mt = w & 3, wc = w >> 2;
    int m0 = blockIdx.x * 64 + mt * 16;
    int h = lane >> 4, l15 = lane & 15;

    const f16* arow = A + (size_t)(m0 + l15) * lda + 8 * h;

    v8f acc[8] = {};

    for (int t = 0; t < Kt; ++t) {
        int kb = t * 32;
        // A 16x32 fragment: halves 0..7 = K in [kb+8h, kb+8h+8),
        //                   halves 8..15 = K in [kb+16+8h, kb+16+8h+8)
        v8h alo = *(const v8h*)(arow + kb);
        v8h ahi = *(const v8h*)(arow + kb + 16);
        v16h a = __builtin_shufflevector(alo, ahi,
                 0, 1, 2, 3, 4, 5, 6, 7, 8, 9, 10, 11, 12, 13, 14, 15);
        const f16* bbase = Bp + ((size_t)(t * 16 + wc * 8) * 32 + lane) * 16;
#pragma unroll
        for (int jj = 0; jj < 8; ++jj) {
            v16h bf = *(const v16h*)(bbase + (size_t)jj * 32 * 16);
            acc[jj] = __builtin_amdgcn_wmma_f32_16x16x32_f16(
                false, a, false, bf, (short)0, acc[jj], false, false);
        }
    }

#pragma unroll
    for (int jj = 0; jj < 8; ++jj) {
        int n = (wc * 8 + jj) * 16 + l15;
        float bv = bias[n];
#pragma unroll
        for (int r = 0; r < 8; ++r) {
            int rowo = m0 + r + 8 * h;          // C/D: VGPR r -> row r + 8h
            float v = acc[jj][r] + bv;
            if (mode == 0 || mode == 1)      v = v > 0.0f ? v : 0.0f;
            else if (mode == 2)              v = v > 0.0f ? v : 0.01f * v;
            size_t oi = (size_t)rowo * 256 + n;
            if (mode == 1 || mode == 2) ((f16*)outp)[oi] = (f16)v;
            else                        ((float*)outp)[oi] = v;
        }
    }
}

// ---------------------------------------------------------------------------
// Instance norm over 256 (one row per block), f32 in -> f16 out.
// ---------------------------------------------------------------------------
__global__ __launch_bounds__(256) void inorm_kernel(
    const float* __restrict__ in, f16* __restrict__ out)
{
    int row = blockIdx.x, tid = threadIdx.x;
    __shared__ float red[256];
    __shared__ float s_mean, s_rstd;
    float x = in[(size_t)row * 256 + tid];
    red[tid] = x; __syncthreads();
    for (int off = 128; off > 0; off >>= 1) {
        if (tid < off) red[tid] += red[tid + off];
        __syncthreads();
    }
    if (tid == 0) s_mean = red[0] * (1.0f / 256.0f);
    __syncthreads();
    float d = x - s_mean;
    red[tid] = d * d; __syncthreads();
    for (int off = 128; off > 0; off >>= 1) {
        if (tid < off) red[tid] += red[tid + off];
        __syncthreads();
    }
    if (tid == 0) s_rstd = rsqrtf(red[0] * (1.0f / 256.0f) + 1e-5f);
    __syncthreads();
    out[(size_t)row * 256 + tid] = (f16)(d * s_rstd);
}

// ---------------------------------------------------------------------------
// relu -> instance norm -> dot(score_W) + score_b, one row per block.
// ---------------------------------------------------------------------------
__global__ __launch_bounds__(256) void inorm_score_kernel(
    const float* __restrict__ in, const float* __restrict__ sW,
    const float* __restrict__ sb, float* __restrict__ scores)
{
    int row = blockIdx.x, tid = threadIdx.x;
    __shared__ float red[256];
    __shared__ float s_mean, s_rstd;
    float x = in[(size_t)row * 256 + tid];
    x = x > 0.0f ? x : 0.0f;            // leaky_relu then relu == relu
    red[tid] = x; __syncthreads();
    for (int off = 128; off > 0; off >>= 1) {
        if (tid < off) red[tid] += red[tid + off];
        __syncthreads();
    }
    if (tid == 0) s_mean = red[0] * (1.0f / 256.0f);
    __syncthreads();
    float d = x - s_mean;
    red[tid] = d * d; __syncthreads();
    for (int off = 128; off > 0; off >>= 1) {
        if (tid < off) red[tid] += red[tid + off];
        __syncthreads();
    }
    if (tid == 0) s_rstd = rsqrtf(red[0] * (1.0f / 256.0f) + 1e-5f);
    __syncthreads();
    red[tid] = (d * s_rstd) * sW[tid]; __syncthreads();
    for (int off = 128; off > 0; off >>= 1) {
        if (tid < off) red[tid] += red[tid + off];
        __syncthreads();
    }
    if (tid == 0) scores[row] = red[0] + sb[0];
}

// ---------------------------------------------------------------------------
// Softmax over S=4096 per batch (one block per batch).
// ---------------------------------------------------------------------------
__global__ __launch_bounds__(256) void softmax_kernel(
    const float* __restrict__ scores, float* __restrict__ probs)
{
    int b = blockIdx.x, tid = threadIdx.x;
    __shared__ float red[256];
    __shared__ float s_max, s_sum;
    float mx = -3.4e38f;
    for (int i = tid; i < NS; i += 256) mx = fmaxf(mx, scores[b * NS + i]);
    red[tid] = mx; __syncthreads();
    for (int off = 128; off > 0; off >>= 1) {
        if (tid < off) red[tid] = fmaxf(red[tid], red[tid + off]);
        __syncthreads();
    }
    if (tid == 0) s_max = red[0];
    __syncthreads();
    float sm = 0.0f;
    for (int i = tid; i < NS; i += 256) sm += expf(scores[b * NS + i] - s_max);
    red[tid] = sm; __syncthreads();
    for (int off = 128; off > 0; off >>= 1) {
        if (tid < off) red[tid] += red[tid + off];
        __syncthreads();
    }
    if (tid == 0) s_sum = red[0];
    __syncthreads();
    float inv = 1.0f / s_sum;
    for (int i = tid; i < NS; i += 256)
        probs[b * NS + i] = expf(scores[b * NS + i] - s_max) * inv;
}

__global__ void zero_kernel(float* __restrict__ p, int n) {
    int i = blockIdx.x * 256 + threadIdx.x;
    if (i < n) p[i] = 0.0f;
}

// ---------------------------------------------------------------------------
// Scatter-add: spect[b][mass_idx[b,s,p]] += intensity[b,s,p] * probs[b,s]
// ---------------------------------------------------------------------------
__global__ void scatter_kernel(const int* __restrict__ mass_idx,
                               const float* __restrict__ intensity,
                               const float* __restrict__ probs,
                               float* __restrict__ spect)
{
    int idx = blockIdx.x * 256 + threadIdx.x;
    if (idx >= NB * NS * NP) return;
    int b = idx / (NS * NP);
    int s = (idx / NP) % NS;
    float w = intensity[idx] * probs[b * NS + s];
    int bin = mass_idx[idx];
    __hip_atomic_fetch_add(&spect[(size_t)b * NBINS + bin], w,
                           __ATOMIC_RELAXED, __HIP_MEMORY_SCOPE_AGENT);
}

// ---------------------------------------------------------------------------
extern "C" void kernel_launch(void* const* d_in, const int* in_sizes, int n_in,
                              void* d_out, int out_size, void* d_ws, size_t ws_size,
                              hipStream_t stream) {
    const float* vert_feat = (const float*)d_in[0];
    const float* vert_mask = (const float*)d_in[1];
    const float* eloh      = (const float*)d_in[2];
    const int*   subsets   = (const int*)d_in[4];
    const int*   mass_idx  = (const int*)d_in[6];
    const float* intensity = (const float*)d_in[7];
    const float* combine_W = (const float*)d_in[8];
    const float* combine_b = (const float*)d_in[9];
    const float* l1_W  = (const float*)d_in[10];
    const float* l1_b  = (const float*)d_in[11];
    const float* l2a_W = (const float*)d_in[12];
    const float* l2a_b = (const float*)d_in[13];
    const float* l2b_W = (const float*)d_in[14];
    const float* l2b_b = (const float*)d_in[15];
    const float* score_W = (const float*)d_in[16];
    const float* score_b = (const float*)d_in[17];

    float* spect = (float*)d_out;                       // (B, NBINS)
    float* probs = (float*)d_out + (size_t)NB * NBINS;  // (B, S)

    char* ws = (char*)d_ws;
    f16*   pack_c   = (f16*)(ws + 0);                      // 12*8192 halfs
    f16*   pack_l1  = (f16*)(ws + 196608);
    f16*   pack_l2a = (f16*)(ws + 327680);
    f16*   pack_l2b = (f16*)(ws + 458752);
    f16*   act0     = (f16*)(ws + 589824);                 // ROWS x 384 f16
    float* zbuf     = (float*)(ws + 25755648);             // ROWS x 256 f32
    f16*   actA     = (f16*)(ws + 59310080);               // ROWS x 256 f16
    f16*   actB     = (f16*)(ws + 76087296);               // ROWS x 256 f16
    float* scores   = (float*)(ws + 92864512);             // ROWS f32

    // 1) weight repack to WMMA fragment layout
    pack_w_kernel<<<(12 * 8192 + 255) / 256, 256, 0, stream>>>(combine_W, 356, KPAD, pack_c);
    pack_w_kernel<<<(8 * 8192 + 255) / 256, 256, 0, stream>>>(l1_W,  256, 256, pack_l1);
    pack_w_kernel<<<(8 * 8192 + 255) / 256, 256, 0, stream>>>(l2a_W, 256, 256, pack_l2a);
    pack_w_kernel<<<(8 * 8192 + 255) / 256, 256, 0, stream>>>(l2b_W, 256, 256, pack_l2b);

    // 2) subset features -> padded activation rows
    stage_a_kernel<<<ROWS, 256, 0, stream>>>(vert_feat, vert_mask, eloh, subsets, act0);

    // 3) MLP chain via WMMA
    gemm_wmma_kernel<<<ROWS / 64, 256, 0, stream>>>(act0, KPAD, pack_c, 12,
                                                    combine_b, zbuf, 0);     // relu -> f32
    inorm_kernel<<<ROWS, 256, 0, stream>>>(zbuf, actA);                      // norm_post_combine
    gemm_wmma_kernel<<<ROWS / 64, 256, 0, stream>>>(actA, 256, pack_l1, 8,
                                                    l1_b, actB, 1);          // relu -> f16
    gemm_wmma_kernel<<<ROWS / 64, 256, 0, stream>>>(actB, 256, pack_l2a, 8,
                                                    l2a_b, actA, 2);         // leaky -> f16
    gemm_wmma_kernel<<<ROWS / 64, 256, 0, stream>>>(actA, 256, pack_l2b, 8,
                                                    l2b_b, zbuf, 3);         // raw -> f32
    inorm_score_kernel<<<ROWS, 256, 0, stream>>>(zbuf, score_W, score_b, scores);

    // 4) softmax + spectral binning
    softmax_kernel<<<NB, 256, 0, stream>>>(scores, probs);
    zero_kernel<<<(NB * NBINS + 255) / 256, 256, 0, stream>>>(spect, NB * NBINS);
    scatter_kernel<<<(NB * NS * NP + 255) / 256, 256, 0, stream>>>(mass_idx, intensity,
                                                                   probs, spect);
}
